// LiquidNeuralNetwork_89395449299101
// MI455X (gfx1250) — matrix operand order repaired and checked by reading, not compiled, not run on hardware
//
#include <hip/hip_runtime.h>
#include <hip/hip_bf16.h>
#include <math.h>

// Problem constants (from reference)
#define B_ 256
#define S_ 64
#define I_ 256
#define H_ 1024
#define O_ 10

typedef __bf16 v16bf __attribute__((ext_vector_type(16)));
typedef __bf16 v8bf  __attribute__((ext_vector_type(8)));
typedef float  v8f   __attribute__((ext_vector_type(8)));

// Branch-free hardware tanh: tanh(x) = 1 - 2/(exp(2x)+1)
// v_exp_f32 + v_rcp_f32 are single TRANS ops; saturates correctly at +/-inf.
__device__ __forceinline__ float tanh_fast(float x) {
    float e = __builtin_amdgcn_exp2f(x * 2.88539008177793f); // exp(2x)
    return 1.0f - 2.0f * __builtin_amdgcn_rcpf(e + 1.0f);
}

// ---------------------------------------------------------------------------
// Phase 0: f32 -> bf16 conversion (X and W_fc), one-time, bandwidth-trivial.
// ---------------------------------------------------------------------------
__global__ void cvt_f32_to_bf16(const float* __restrict__ src,
                                __bf16* __restrict__ dst, int n) {
    int i = blockIdx.x * blockDim.x + threadIdx.x;
    if (i < n) dst[i] = (__bf16)src[i];
}

// Load one 16x256 A tile (8 K-chunks) for this lane into distinct registers.
// A 16x32 bf16 layout: lane holds row M = lane&15; elems 0..7 -> K = 8*half+e,
// elems 8..15 -> K = 16 + 8*half + e (within each 32-wide chunk).
__device__ __forceinline__ void load_a_tile(v16bf a[8],
                                            const __bf16* __restrict__ arow,
                                            int halfOff8) {
    #pragma unroll
    for (int kc = 0; kc < 8; ++kc) {
        const v8bf* p = (const v8bf*)(arow + kc * 32 + halfOff8);
        v8bf lo = p[0];   // 16 bytes
        v8bf hi = p[2];   // +16 elements = +32 bytes
        a[kc] = __builtin_shufflevector(lo, hi, 0, 1, 2, 3, 4, 5, 6, 7,
                                        8, 9, 10, 11, 12, 13, 14, 15);
    }
}

// ---------------------------------------------------------------------------
// Phase 1: fused GEMM + ODE integration, software-pipelined.
// One wave owns a 16(B) x 16(H) tile. W fragments (64 VGPRs) and h (8 VGPRs)
// stay resident for all 64 steps; the A tile is double-buffered so next-step
// global loads overlap the current 8-WMMA chain + RK4 VALU work.
// ---------------------------------------------------------------------------
__global__ __launch_bounds__(32) void lnn_integrate(
    const __bf16* __restrict__ Xbf,   // [B, S, I] row-major in I
    const __bf16* __restrict__ Wbf,   // [H, I]   row-major in I
    const float*  __restrict__ b_fc,  // [H]
    const float*  __restrict__ alpha, // [H]
    const float*  __restrict__ beta,  // [H]
    const float*  __restrict__ gamma, // [H]
    float*        __restrict__ Hout)  // [B, H]
{
    const int lane = threadIdx.x;     // 0..31 (wave32)
    const int half = lane >> 4;       // 0 or 1
    const int l16  = lane & 15;
    const int tileH = blockIdx.x * 16;   // 64 tiles
    const int tileB = blockIdx.y * 16;   // 16 tiles

    // --- B fragments (W_fc^T tile), resident across all steps ---
    // B 32x16 bf16: lane holds column n=l16; elems 0..7 -> K = 8*half+e, elems 8..15 -> K = 16 + 8*half + e.
    v16bf wfrag[8];
    {
        const __bf16* wrow = Wbf + (size_t)(tileH + l16) * I_ + half * 8;
        #pragma unroll
        for (int kc = 0; kc < 8; ++kc) {
            const v8bf* p = (const v8bf*)(wrow + kc * 32);
            v8bf lo = p[0];
            v8bf hi = p[2];
            wfrag[kc] = __builtin_shufflevector(lo, hi, 0, 1, 2, 3, 4, 5, 6, 7,
                                                8, 9, 10, 11, 12, 13, 14, 15);
        }
    }

    // Per-column parameters (column n = tileH + l16, same for all 8 C vgprs)
    const int ncol = tileH + l16;
    const float bf = b_fc[ncol];
    const float al = alpha[ncol];
    const float be = beta[ncol];
    const float ga = gamma[ncol];

    // h accumulator in C/D layout: vgpr v, lane -> (m = v + 8*half, n = l16)
    v8f h;
    #pragma unroll
    for (int v = 0; v < 8; ++v) h[v] = 0.0f;

    const __bf16* xrow = Xbf + (size_t)(tileB + l16) * S_ * I_;
    const int halfOff8 = half * 8;
    const float dt = (1.0f / 64.0f) * 0.5f;   // 2 RK4 substeps per interval

    // One integration step: u-tile = WMMA chain over a[8]; then RK4 per elem.
    auto step = [&](const v16bf a[8]) {
        v8f acc;
        #pragma unroll
        for (int v = 0; v < 8; ++v) acc[v] = bf;
        #pragma unroll
        for (int kc = 0; kc < 8; ++kc) {
            acc = __builtin_amdgcn_wmma_f32_16x16x32_bf16(
                false, a[kc], false, wfrag[kc], (short)0, acc, false, false);
        }
        #pragma unroll
        for (int v = 0; v < 8; ++v) {
            float u  = acc[v];
            float hv = h[v];
            #pragma unroll
            for (int sub = 0; sub < 2; ++sub) {
                float k1 = -al * hv + be * tanh_fast(u + ga * hv);
                float h2 = hv + 0.5f * dt * k1;
                float k2 = -al * h2 + be * tanh_fast(u + ga * h2);
                float h3 = hv + 0.5f * dt * k2;
                float k3 = -al * h3 + be * tanh_fast(u + ga * h3);
                float h4 = hv + dt * k3;
                float k4 = -al * h4 + be * tanh_fast(u + ga * h4);
                hv += (dt * (1.0f / 6.0f)) * (k1 + 2.0f * k2 + 2.0f * k3 + k4);
            }
            h[v] = hv;
        }
    };

    // Software pipeline: double-buffered A tile, s unrolled by 2.
    v16bf a0[8], a1[8];
    load_a_tile(a0, xrow, halfOff8);
    for (int s = 0; s < S_; s += 2) {
        load_a_tile(a1, xrow + (size_t)(s + 1) * I_, halfOff8);  // s+1 <= 63
        step(a0);
        int snext = (s + 2 < S_) ? (s + 2) : (S_ - 1);           // clamp tail
        load_a_tile(a0, xrow + (size_t)snext * I_, halfOff8);
        step(a1);
    }

    // ---- store final h tile (C/D layout -> row-major [B,H]) ----
    float* orow = Hout + (size_t)tileB * H_ + tileH;
    #pragma unroll
    for (int v = 0; v < 8; ++v) {
        int m = v + 8 * half;
        orow[(size_t)m * H_ + l16] = h[v];
    }
}

// ---------------------------------------------------------------------------
// Phase 2: tiny output projection out[b,o] = h[b,:] . W_out[o,:] + b_out[o]
// ---------------------------------------------------------------------------
__global__ void lnn_out(const float* __restrict__ Hf,
                        const float* __restrict__ Wout,
                        const float* __restrict__ bout,
                        float* __restrict__ out) {
    int idx = blockIdx.x * blockDim.x + threadIdx.x;
    if (idx >= B_ * O_) return;
    int b = idx / O_;
    int o = idx - b * O_;
    const float* hr = Hf + (size_t)b * H_;
    const float* wr = Wout + (size_t)o * H_;
    float s = bout[o];
    #pragma unroll 4
    for (int i = 0; i < H_; i += 4) {
        s += hr[i] * wr[i] + hr[i + 1] * wr[i + 1] +
             hr[i + 2] * wr[i + 2] + hr[i + 3] * wr[i + 3];
    }
    out[idx] = s;
}

// ---------------------------------------------------------------------------
// Launch.  Inputs: x, W_fc, b_fc, alpha, beta, gamma, W_out, b_out
// ---------------------------------------------------------------------------
extern "C" void kernel_launch(void* const* d_in, const int* in_sizes, int n_in,
                              void* d_out, int out_size, void* d_ws, size_t ws_size,
                              hipStream_t stream) {
    const float* x     = (const float*)d_in[0];   // [B,S,I]
    const float* W_fc  = (const float*)d_in[1];   // [H,I]
    const float* b_fc  = (const float*)d_in[2];   // [H]
    const float* alpha = (const float*)d_in[3];   // [H]
    const float* beta  = (const float*)d_in[4];   // [H]
    const float* gamma = (const float*)d_in[5];   // [H]
    const float* W_out = (const float*)d_in[6];   // [O,H]
    const float* b_out = (const float*)d_in[7];   // [O]
    float* out = (float*)d_out;                   // [B,O]

    // Workspace layout (256B aligned): Xbf | Wbf | Hout  (~9.9 MB total)
    char* ws = (char*)d_ws;
    const size_t xbf_bytes = (size_t)B_ * S_ * I_ * sizeof(__bf16);
    const size_t wbf_bytes = (size_t)H_ * I_ * sizeof(__bf16);
    __bf16* Xbf  = (__bf16*)ws;
    __bf16* Wbf  = (__bf16*)(ws + ((xbf_bytes + 255) & ~(size_t)255));
    float*  Hout = (float*)(ws + ((xbf_bytes + 255) & ~(size_t)255)
                               + ((wbf_bytes + 255) & ~(size_t)255));

    // Phase 0: convert X and W_fc to bf16
    {
        int nx = B_ * S_ * I_;
        cvt_f32_to_bf16<<<(nx + 255) / 256, 256, 0, stream>>>(x, Xbf, nx);
        int nw = H_ * I_;
        cvt_f32_to_bf16<<<(nw + 255) / 256, 256, 0, stream>>>(W_fc, Wbf, nw);
    }

    // Phase 1: fused WMMA-GEMM + RK4 integration. One wave per 16x16 tile.
    {
        dim3 grid(H_ / 16, B_ / 16);  // (64, 16)
        lnn_integrate<<<grid, 32, 0, stream>>>(Xbf, Wbf, b_fc, alpha, beta,
                                               gamma, Hout);
    }

    // Phase 2: output projection
    {
        int n = B_ * O_;
        lnn_out<<<(n + 127) / 128, 128, 0, stream>>>(Hout, W_out, b_out, out);
    }
}
